// MossAudioTokenizerLFQ_46849503264984
// MI455X (gfx1250) — matrix operand order repaired and checked by reading, not compiled, not run on hardware
//
#include <hip/hip_runtime.h>
#include <hip/hip_bf16.h>
#include <math.h>

typedef __attribute__((ext_vector_type(2)))  float  v2f;
typedef __attribute__((ext_vector_type(8)))  float  v8f;
typedef __attribute__((ext_vector_type(8)))  __bf16 v8bf;
typedef __attribute__((ext_vector_type(16))) __bf16 v16bf;

static constexpr int NB = 4;      // batch
static constexpr int NC = 1024;   // channels
static constexpr int NT = 4096;   // time
static constexpr int ND = 256;    // latent dim
static constexpr int NK = 8192;   // codebook entries
#define EPSN 1e-12f

// ---------------------------------------------------------------------------
// CDNA5 async global->LDS copy (ASYNCcnt path), with synchronous fallback.
// Builtin signature (from compiler diagnostic): param0 is
// 'int __attribute__((vector_size(16))) __device__(AS1) *'.
// ---------------------------------------------------------------------------
#if defined(__gfx1250__) && __has_builtin(__builtin_amdgcn_global_load_async_to_lds_b128) && \
    __has_builtin(__builtin_amdgcn_s_wait_asynccnt)
#define USE_ASYNC_LDS 1
#else
#define USE_ASYNC_LDS 0
#endif

#if USE_ASYNC_LDS
typedef int v4i_vs __attribute__((vector_size(16)));
typedef __attribute__((address_space(1))) v4i_vs* gv4i_t;
typedef __attribute__((address_space(3))) v4i_vs* lv4i_t;

__device__ __forceinline__ void async_b128(void* lds_dst, const void* gsrc) {
  __builtin_amdgcn_global_load_async_to_lds_b128(
      (gv4i_t)(uintptr_t)gsrc,
      (lv4i_t)(unsigned)(uintptr_t)lds_dst,  // generic LDS addr: low 32 bits = LDS offset
      0, 0);
}
__device__ __forceinline__ void async_wait0() {
  __builtin_amdgcn_s_wait_asynccnt(0);
}
#else
__device__ __forceinline__ void async_b128(void* lds_dst, const void* gsrc) {
  *(float4*)lds_dst = *(const float4*)gsrc;
}
__device__ __forceinline__ void async_wait0() {}
#endif

// ---------------------------------------------------------------------------
// Weight-norm scales: scale[r] = g[r] / ||v[r,:]||   (one wave per row)
// ---------------------------------------------------------------------------
__global__ void __launch_bounds__(256) k_row_scale(
    const float* __restrict__ v, const float* __restrict__ g,
    float* __restrict__ scale, int rows, int cols) {
  int wid  = (blockIdx.x * blockDim.x + threadIdx.x) >> 5;
  int lane = threadIdx.x & 31;
  if (wid >= rows) return;
  const float* row = v + (size_t)wid * cols;
  float s = 0.f;
  for (int i = lane; i < cols; i += 32) { float x = row[i]; s += x * x; }
#pragma unroll
  for (int off = 16; off; off >>= 1) s += __shfl_xor(s, off, 32);
  if (lane == 0) scale[wid] = g[wid] / sqrtf(s);
}

// ---------------------------------------------------------------------------
// Codebook prep: normalized bf16 copy + ||cb_n||^2 per entry (one wave/row)
// ---------------------------------------------------------------------------
__global__ void __launch_bounds__(256) k_cb_prep(
    const float* __restrict__ cb, float* __restrict__ cn2,
    __bf16* __restrict__ cbn) {
  int wid  = (blockIdx.x * blockDim.x + threadIdx.x) >> 5;
  int lane = threadIdx.x & 31;
  if (wid >= NK) return;
  const float* row = cb + (size_t)wid * ND;
  float s = 0.f;
#pragma unroll
  for (int i = 0; i < ND / 32; ++i) { float x = row[lane + 32 * i]; s += x * x; }
#pragma unroll
  for (int off = 16; off; off >>= 1) s += __shfl_xor(s, off, 32);
  float inv = 1.0f / fmaxf(sqrtf(s), EPSN);
  __bf16* dst = cbn + (size_t)wid * ND;
#pragma unroll
  for (int i = 0; i < ND / 32; ++i)
    dst[lane + 32 * i] = (__bf16)(row[lane + 32 * i] * inv);
  if (lane == 0) cn2[wid] = s * inv * inv;
}

// ---------------------------------------------------------------------------
// in-proj GEMM (fp32 WMMA 16x16x4):  z_e[b,d,t] = s_in[d]*(in_v[d,:].z[b,:,t]) + in_b[d]
// Block tile 128(M=d) x 128(N=t), K chunks of 32; 8 waves as 4x2, wave tile 32x64.
// ---------------------------------------------------------------------------
__global__ void __launch_bounds__(256) k_in_proj(
    const float* __restrict__ in_v,   // [ND, NC]
    const float* __restrict__ z,      // [NB, NC, NT]
    const float* __restrict__ scale_in, const float* __restrict__ in_b,
    float* __restrict__ z_e)          // [NB, ND, NT]
{
  __shared__ float As[128][36];   // 144B row stride (16B multiple)
  __shared__ float Bs[32][136];   // 544B row stride (16B multiple)
  const int tid  = threadIdx.x;
  const int lane = tid & 31;
  const int wid  = tid >> 5;
  const int l15  = lane & 15, hl = lane >> 4;
  const int wm = wid >> 1, wn = wid & 1;
  const int b     = blockIdx.z;
  const int mbase = blockIdx.y * 128;
  const int tbase = blockIdx.x * 128;

  v8f acc[2][4] = {};

  const int a_kq = tid & 7,  a_m = tid >> 3;
  const int b_tq = tid & 31, b_k = tid >> 5;

  for (int kc = 0; kc < NC; kc += 32) {
    __syncthreads();
#pragma unroll
    for (int r = 0; r < 4; ++r) {
      int m = a_m + 32 * r;
      async_b128(&As[m][4 * a_kq],
                 &in_v[(size_t)(mbase + m) * NC + kc + 4 * a_kq]);
    }
#pragma unroll
    for (int r = 0; r < 4; ++r) {
      int k = b_k + 8 * r;
      async_b128(&Bs[k][4 * b_tq],
                 &z[((size_t)b * NC + kc + k) * NT + tbase + 4 * b_tq]);
    }
    async_wait0();
    __syncthreads();
#pragma unroll
    for (int ks = 0; ks < 32; ks += 4) {
      v2f a[2], bb[4];
#pragma unroll
      for (int mt = 0; mt < 2; ++mt)
        a[mt] = *(const v2f*)&As[wm * 32 + mt * 16 + l15][ks + 2 * hl];
#pragma unroll
      for (int nt = 0; nt < 4; ++nt) {
        int n = wn * 64 + nt * 16 + l15;
        v2f bv; bv[0] = Bs[ks + 2 * hl + 0][n]; bv[1] = Bs[ks + 2 * hl + 1][n];
        bb[nt] = bv;
      }
#pragma unroll
      for (int mt = 0; mt < 2; ++mt)
#pragma unroll
        for (int nt = 0; nt < 4; ++nt)
          acc[mt][nt] = __builtin_amdgcn_wmma_f32_16x16x4_f32(
              false, a[mt], false, bb[nt], (short)0, acc[mt][nt], false, false);
    }
  }
#pragma unroll
  for (int mt = 0; mt < 2; ++mt) {
#pragma unroll
    for (int r = 0; r < 8; ++r) {
      int d = mbase + wm * 32 + mt * 16 + hl * 8 + r;
      float sc = scale_in[d], bi = in_b[d];
#pragma unroll
      for (int nt = 0; nt < 4; ++nt) {
        int t = tbase + wn * 64 + nt * 16 + l15;
        z_e[((size_t)b * ND + d) * NT + t] = acc[mt][nt][r] * sc + bi;
      }
    }
  }
}

// ---------------------------------------------------------------------------
// enc normalization: encn[n,d] = bf16(z_e[b,d,t] / max(||row||,eps)), n=b*T+t
// ---------------------------------------------------------------------------
__global__ void __launch_bounds__(256) k_enc_norm(
    const float* __restrict__ z_e, __bf16* __restrict__ encn) {
  int b = blockIdx.y;
  int tbase = blockIdx.x * 64;
  int tl = threadIdx.x & 63;
  int dg = threadIdx.x >> 6;  // 0..3
  const float* base = z_e + (size_t)b * ND * NT + tbase + tl;
  float s = 0.f;
  for (int d = dg * 64; d < dg * 64 + 64; ++d) {
    float v = base[(size_t)d * NT];
    s += v * v;
  }
  __shared__ float red[4][64];
  red[dg][tl] = s;
  __syncthreads();
  float tot = red[0][tl] + red[1][tl] + red[2][tl] + red[3][tl];
  float inv = 1.0f / fmaxf(sqrtf(tot), EPSN);
  size_t n = (size_t)b * NT + tbase + tl;
  __bf16* dst = encn + n * ND;
  for (int d = dg * 64; d < dg * 64 + 64; ++d)
    dst[d] = (__bf16)(base[(size_t)d * NT] * inv);
}

// ---------------------------------------------------------------------------
// Fused distance GEMM + argmax (bf16 WMMA 16x16x32, fp32 accum).
// score(n,k) = 2*enc_n(n).cb_n(k) - ||cb_n(k)||^2  ; argmax over k.
// Wave owns 16 enc rows (B fragments resident in VGPRs, 64 regs);
// codebook streamed via LDS in 64-entry chunks (A fragments, async copy).
// ---------------------------------------------------------------------------
__global__ void __launch_bounds__(256) k_vq_argmin(
    const __bf16* __restrict__ encn, const __bf16* __restrict__ cbn,
    const float* __restrict__ cn2, int* __restrict__ idx_out) {
  __shared__ __align__(32) __bf16 cbs[64 * ND];
  __shared__ float cn2s[64];
  const int tid  = threadIdx.x;
  const int lane = tid & 31, wid = tid >> 5;
  const int l15  = lane & 15, hl = lane >> 4;
  const int rowbase = blockIdx.x * 128 + wid * 16;

  // resident B fragments: 8 K-chunks of 32 for this wave's 16 rows
  v16bf bfrag[8];
  {
    const __bf16* rp = encn + (size_t)(rowbase + l15) * ND;
#pragma unroll
    for (int kk = 0; kk < 8; ++kk)
      bfrag[kk] = *(const v16bf*)(rp + kk * 32 + hl * 16);
  }

  float best = -3.0e38f;
  int   bidx = 0;

  for (int kc = 0; kc < NK; kc += 64) {
    __syncthreads();
    {  // stage 64 codebook rows (bf16) + their cn2
      int row = tid >> 2, seg = tid & 3;
      const __bf16* src = cbn + (size_t)(kc + row) * ND + seg * 64;
      __bf16* dst = cbs + row * ND + seg * 64;
#pragma unroll
      for (int j = 0; j < 8; ++j) async_b128(dst + j * 8, src + j * 8);
      if (tid < 64) cn2s[tid] = cn2[kc + tid];
    }
    async_wait0();
    __syncthreads();
#pragma unroll
    for (int mt = 0; mt < 4; ++mt) {
      v8f acc = {};
      const __bf16* ap = cbs + (size_t)(mt * 16 + l15) * ND;
#pragma unroll
      for (int kk = 0; kk < 8; ++kk) {
        v8bf lo = *(const v8bf*)(ap + kk * 32 + hl * 16);
        v8bf hi = *(const v8bf*)(ap + kk * 32 + hl * 16 + 8);
        v16bf af = __builtin_shufflevector(lo, hi, 0, 1, 2, 3, 4, 5, 6, 7, 8, 9,
                                           10, 11, 12, 13, 14, 15);
        acc = __builtin_amdgcn_wmma_f32_16x16x32_bf16(
            false, af, false, bfrag[kk], (short)0, acc, false, false);
      }
#pragma unroll
      for (int r = 0; r < 8; ++r) {
        int mloc = mt * 16 + hl * 8 + r;
        float sc = 2.0f * acc[r] - cn2s[mloc];
        int gk = kc + mloc;
        if (sc > best) { best = sc; bidx = gk; }  // ascending gk: first-max kept
      }
    }
  }
  // combine lane halves (same enc row, disjoint codeword subsets)
  float ob = __shfl_xor(best, 16, 32);
  int   oi = __shfl_xor(bidx, 16, 32);
  if (ob > best || (ob == best && oi < bidx)) { best = ob; bidx = oi; }
  if (hl == 0) idx_out[rowbase + l15] = bidx;
}

// ---------------------------------------------------------------------------
// out-proj GEMM (fp32 WMMA) with codebook gather:
// out[b,c,t] = s_out[c]*(out_v[c,:].cb[idx[b,t],:]) + out_b[c]
// ---------------------------------------------------------------------------
__global__ void __launch_bounds__(256) k_out_proj(
    const float* __restrict__ out_v,     // [NC, ND]
    const float* __restrict__ codebook,  // [NK, ND]
    const int* __restrict__ idx,         // [NB, NT]
    const float* __restrict__ scale_out, const float* __restrict__ out_b,
    float* __restrict__ outp)            // [NB, NC, NT]
{
  __shared__ float As[128][36];
  __shared__ float Bs[32][136];
  __shared__ int idxs[128];
  const int tid  = threadIdx.x;
  const int lane = tid & 31, wid = tid >> 5;
  const int l15  = lane & 15, hl = lane >> 4;
  const int wm = wid >> 1, wn = wid & 1;
  const int b     = blockIdx.z;
  const int cbase = blockIdx.y * 128;
  const int tbase = blockIdx.x * 128;

  if (tid < 128) idxs[tid] = idx[(size_t)b * NT + tbase + tid];

  v8f acc[2][4] = {};
  const int a_kq = tid & 7, a_m = tid >> 3;

  for (int kc = 0; kc < ND; kc += 32) {
    __syncthreads();
#pragma unroll
    for (int r = 0; r < 4; ++r) {
      int m = a_m + 32 * r;
      async_b128(&As[m][4 * a_kq],
                 &out_v[(size_t)(cbase + m) * ND + kc + 4 * a_kq]);
    }
    {  // gather z_q columns from codebook (scatter into LDS -> synchronous)
      int col = tid & 127, hk = tid >> 7;
      const float* src = codebook + (size_t)idxs[col] * ND + kc + hk * 16;
#pragma unroll
      for (int j = 0; j < 16; ++j) Bs[hk * 16 + j][col] = src[j];
    }
    async_wait0();
    __syncthreads();
#pragma unroll
    for (int ks = 0; ks < 32; ks += 4) {
      v2f a[2], bb[4];
#pragma unroll
      for (int mt = 0; mt < 2; ++mt)
        a[mt] = *(const v2f*)&As[wm * 32 + mt * 16 + l15][ks + 2 * hl];
#pragma unroll
      for (int nt = 0; nt < 4; ++nt) {
        int n = wn * 64 + nt * 16 + l15;
        v2f bv; bv[0] = Bs[ks + 2 * hl + 0][n]; bv[1] = Bs[ks + 2 * hl + 1][n];
        bb[nt] = bv;
      }
#pragma unroll
      for (int mt = 0; mt < 2; ++mt)
#pragma unroll
        for (int nt = 0; nt < 4; ++nt)
          acc[mt][nt] = __builtin_amdgcn_wmma_f32_16x16x4_f32(
              false, a[mt], false, bb[nt], (short)0, acc[mt][nt], false, false);
    }
  }
#pragma unroll
  for (int mt = 0; mt < 2; ++mt) {
#pragma unroll
    for (int r = 0; r < 8; ++r) {
      int c = cbase + wm * 32 + mt * 16 + hl * 8 + r;
      float sc = scale_out[c], bi = out_b[c];
#pragma unroll
      for (int nt = 0; nt < 4; ++nt) {
        int t = tbase + wn * 64 + nt * 16 + l15;
        outp[((size_t)b * NC + c) * NT + t] = acc[mt][nt][r] * sc + bi;
      }
    }
  }
}

// ---------------------------------------------------------------------------
extern "C" void kernel_launch(void* const* d_in, const int* in_sizes, int n_in,
                              void* d_out, int out_size, void* d_ws, size_t ws_size,
                              hipStream_t stream) {
  (void)in_sizes; (void)n_in; (void)out_size; (void)ws_size;
  const float* z        = (const float*)d_in[0];
  const float* in_v     = (const float*)d_in[1];
  const float* in_g     = (const float*)d_in[2];
  const float* in_b     = (const float*)d_in[3];
  const float* out_v    = (const float*)d_in[4];
  const float* out_g    = (const float*)d_in[5];
  const float* out_b    = (const float*)d_in[6];
  const float* codebook = (const float*)d_in[7];

  // d_out = [out (B*C*T) | indices (B*T as int32) | z_e (B*D*T)]
  float* outp = (float*)d_out;
  int*   idxo = (int*)(outp + (size_t)NB * NC * NT);
  float* z_e  = outp + (size_t)NB * NC * NT + (size_t)NB * NT;

  // workspace layout
  char* ws = (char*)d_ws;
  float*  scale_in  = (float*)(ws + 0);        // 256 f
  float*  scale_out = (float*)(ws + 1024);     // 1024 f
  float*  cn2       = (float*)(ws + 8192);     // 8192 f
  __bf16* cbn       = (__bf16*)(ws + 65536);                        // 4 MB
  __bf16* encn      = (__bf16*)(ws + 65536 + (size_t)NK * ND * 2);  // 8 MB

  k_row_scale<<<(ND * 32) / 256, 256, 0, stream>>>(in_v, in_g, scale_in, ND, NC);
  k_row_scale<<<(NC * 32) / 256, 256, 0, stream>>>(out_v, out_g, scale_out, NC, ND);
  k_cb_prep<<<(NK * 32) / 256, 256, 0, stream>>>(codebook, cn2, cbn);
  k_in_proj<<<dim3(NT / 128, ND / 128, NB), 256, 0, stream>>>(in_v, z, scale_in,
                                                              in_b, z_e);
  k_enc_norm<<<dim3(NT / 64, NB), 256, 0, stream>>>(z_e, encn);
  k_vq_argmin<<<(NB * NT) / 128, 256, 0, stream>>>(encn, cbn, cn2, idxo);
  k_out_proj<<<dim3(NT / 128, NC / 128, NB), 256, 0, stream>>>(
      out_v, codebook, idxo, scale_out, out_b, outp);
}
